// GPT2Attention_60120952209784
// MI455X (gfx1250) — matrix operand order repaired and checked by reading, not compiled, not run on hardware
//
#include <hip/hip_runtime.h>

// ---------------- CDNA5 WMMA types ----------------
typedef __attribute__((ext_vector_type(16))) __bf16 v16bf;
typedef __attribute__((ext_vector_type(8)))  float  v8f;

#define T_SEQ 4096
#define C_DIM 1024
#define N_HEAD 16
#define HEAD_DIM 64
#define QKV_N (3 * C_DIM)
// 1/sqrt(64) * log2(e): scores computed directly in log2 domain (v_exp_f32 is base-2)
#define ATTN_SCALE_LOG2E (0.125f * 1.4426950408889634f)

__device__ __forceinline__ unsigned short f2bf(float f) {
  unsigned int u = __float_as_uint(f);
  unsigned int r = (u + 0x7FFFu + ((u >> 16) & 1u)) >> 16;
  return (unsigned short)r;
}

__device__ __forceinline__ v8f wmma_bf16(v16bf a, v16bf b, v8f c) {
  // D = A(16x32 bf16) * B(32x16 bf16) + C(16x16 f32)
  return __builtin_amdgcn_wmma_f32_16x16x32_bf16(
      false, a, false, b, (short)0, c, false, false);
}

// A fragment 16x32 (MxK), row-major source [*, ld]:
// lanes 0-15 : M=lane,    K = kbase + {0..7, 16..23}
// lanes 16-31: M=lane-16, K = kbase + {8..15, 24..31}
__device__ __forceinline__ v16bf load_frag_a(const unsigned short* base, int ld,
                                             int mbase, int kbase, int lane) {
  int m  = mbase + (lane & 15);
  int k0 = kbase + ((lane >> 4) << 3);
  const unsigned short* p = base + (size_t)m * ld + k0;
  v16bf f;
  *((uint4*)&f)     = *(const uint4*)(p);
  *((uint4*)&f + 1) = *(const uint4*)(p + 16);
  return f;
}

// B fragment 32x16 (KxN) from transposed storage Bt[N][K] with row stride ld:
// lanes 0-15 : N=lane,    K = kbase + {0..7, 16..23}
// lanes 16-31: N=lane-16, K = kbase + {8..15, 24..31}
__device__ __forceinline__ v16bf load_frag_b(const unsigned short* baseT, int ld,
                                             int nbase, int kbase, int lane) {
  int n  = nbase + (lane & 15);
  int k0 = kbase + ((lane >> 4) << 3);
  const unsigned short* p = baseT + (size_t)n * ld + k0;
  v16bf f;
  *((uint4*)&f)     = *(const uint4*)(p);
  *((uint4*)&f + 1) = *(const uint4*)(p + 16);
  return f;
}

__device__ __forceinline__ uint4 pack8_bf16(v8f a) {
  uint4 o;
  o.x = (unsigned)f2bf(a[0]) | ((unsigned)f2bf(a[1]) << 16);
  o.y = (unsigned)f2bf(a[2]) | ((unsigned)f2bf(a[3]) << 16);
  o.z = (unsigned)f2bf(a[4]) | ((unsigned)f2bf(a[5]) << 16);
  o.w = (unsigned)f2bf(a[6]) | ((unsigned)f2bf(a[7]) << 16);
  return o;
}

// ---------------- conversion kernels ----------------
__global__ void cvt_f32_bf16(const float* __restrict__ src,
                             unsigned short* __restrict__ dst, int n) {
  int i = blockIdx.x * blockDim.x + threadIdx.x;
  if (i < n) dst[i] = f2bf(src[i]);
}

// src [R][Cc] f32 -> dst [Cc][R] bf16
__global__ void tpose_f32_bf16(const float* __restrict__ src,
                               unsigned short* __restrict__ dst, int R, int Cc) {
  int i = blockIdx.x * blockDim.x + threadIdx.x;
  if (i < R * Cc) {
    int r = i / Cc, c = i - r * Cc;
    dst[(size_t)c * R + r] = f2bf(src[i]);
  }
}

// ---------------- QKV projection GEMM ----------------
// 64x64 tile per wave: 16 WMMA per k-step, 16 b128 loads -> 1 load/WMMA.
__global__ __launch_bounds__(128) void qkv_gemm(
    const unsigned short* __restrict__ xh, const unsigned short* __restrict__ WaT,
    const float* __restrict__ b_attn,
    unsigned short* __restrict__ Qs, unsigned short* __restrict__ Ks,
    unsigned short* __restrict__ Vt) {
  const int lane = threadIdx.x & 31;
  const int w    = threadIdx.x >> 5;
  const int ln   = lane & 15;
  const int half = lane >> 4;
  const int ntile = blockIdx.x * 256 + w * 64;   // 64-aligned: region/head uniform
  const int mtile = blockIdx.y * 64;

  v8f acc[4][4];
#pragma unroll
  for (int mi = 0; mi < 4; ++mi)
#pragma unroll
    for (int ni = 0; ni < 4; ++ni) acc[mi][ni] = (v8f){};

  for (int kk = 0; kk < C_DIM; kk += 32) {
    v16bf a[4];
#pragma unroll
    for (int mi = 0; mi < 4; ++mi)
      a[mi] = load_frag_a(xh, C_DIM, mtile + 16 * mi, kk, lane);
#pragma unroll
    for (int ni = 0; ni < 4; ++ni) {
      v16bf b = load_frag_b(WaT, C_DIM, ntile + 16 * ni, kk, lane);
#pragma unroll
      for (int mi = 0; mi < 4; ++mi)
        acc[mi][ni] = wmma_bf16(a[mi], b, acc[mi][ni]);
    }
  }

  const int region = ntile >> 10;          // 0=Q 1=K 2=V (wave-uniform)
  const int h      = (ntile & (C_DIM - 1)) >> 6;   // wave-uniform head
  float bias[4];
#pragma unroll
  for (int ni = 0; ni < 4; ++ni) bias[ni] = b_attn[ntile + 16 * ni + ln];

  if (region == 0) {
#pragma unroll
    for (int mi = 0; mi < 4; ++mi) {
      const int rowbase = mtile + 16 * mi + half * 8;
#pragma unroll
      for (int ni = 0; ni < 4; ++ni) {
        const int d = 16 * ni + ln;
#pragma unroll
        for (int r = 0; r < 8; ++r)
          Qs[((size_t)h * T_SEQ + rowbase + r) * HEAD_DIM + d] =
              f2bf((acc[mi][ni][r] + bias[ni]) * ATTN_SCALE_LOG2E);
      }
    }
  } else if (region == 1) {
#pragma unroll
    for (int mi = 0; mi < 4; ++mi) {
      const int rowbase = mtile + 16 * mi + half * 8;
#pragma unroll
      for (int ni = 0; ni < 4; ++ni) {
        const int d = 16 * ni + ln;
#pragma unroll
        for (int r = 0; r < 8; ++r)
          Ks[((size_t)h * T_SEQ + rowbase + r) * HEAD_DIM + d] =
              f2bf(acc[mi][ni][r] + bias[ni]);
      }
    }
  } else {
#pragma unroll
    for (int mi = 0; mi < 4; ++mi) {
      const int rowbase = mtile + 16 * mi + half * 8;
#pragma unroll
      for (int ni = 0; ni < 4; ++ni) {
        const int d = 16 * ni + ln;
        v8f v;
#pragma unroll
        for (int r = 0; r < 8; ++r) v[r] = acc[mi][ni][r] + bias[ni];
        // transposed V: 8 consecutive t -> one b128 store
        *(uint4*)(Vt + ((size_t)h * HEAD_DIM + d) * T_SEQ + rowbase) = pack8_bf16(v);
      }
    }
  }
}

// ---------------- flash attention tile step ----------------
// 32 query rows x 32 keys: 16 WMMA, 8 fragment loads. MASK only on diagonal tile.
template <bool MASK>
__device__ __forceinline__ void attn_tile(
    const unsigned short* __restrict__ Kh, const unsigned short* __restrict__ Vh,
    unsigned short* __restrict__ Pw, const v16bf (&qa)[2][2], v8f (&o)[2][4],
    float (&m_i)[2][8], float (&l_i)[2][8],
    int kb, int qtile, int lane, int ln, int half) {
  // K fragments: [key-subtile][d-chunk], shared by both row tiles
  v16bf kf[2][2];
  kf[0][0] = load_frag_b(Kh, HEAD_DIM, kb,      0,  lane);
  kf[0][1] = load_frag_b(Kh, HEAD_DIM, kb,      32, lane);
  kf[1][0] = load_frag_b(Kh, HEAD_DIM, kb + 16, 0,  lane);
  kf[1][1] = load_frag_b(Kh, HEAD_DIM, kb + 16, 32, lane);

  v8f sc[2][2];
#pragma unroll
  for (int mi = 0; mi < 2; ++mi)
#pragma unroll
    for (int s = 0; s < 2; ++s) {
      v8f c = {};
      c = wmma_bf16(qa[mi][0], kf[s][0], c);
      c = wmma_bf16(qa[mi][1], kf[s][1], c);
      sc[mi][s] = c;
    }

  const int key0 = kb + ln;
  const int key1 = kb + 16 + ln;
#pragma unroll
  for (int mi = 0; mi < 2; ++mi) {
    float p0[8], p1[8], alpha[8];
#pragma unroll
    for (int r = 0; r < 8; ++r) {
      float s0 = sc[mi][0][r];
      float s1 = sc[mi][1][r];
      if (MASK) {
        const int row = qtile + 16 * mi + r + half * 8;
        s0 = (key0 <= row) ? s0 : -3.0e38f;   // causal mask (diagonal tile only)
        s1 = (key1 <= row) ? s1 : -3.0e38f;
      }
      float mx = fmaxf(s0, s1);
      mx = fmaxf(mx, __shfl_xor(mx, 1));
      mx = fmaxf(mx, __shfl_xor(mx, 2));
      mx = fmaxf(mx, __shfl_xor(mx, 4));
      mx = fmaxf(mx, __shfl_xor(mx, 8));
      float mn = fmaxf(m_i[mi][r], mx);
      alpha[r]   = exp2f(m_i[mi][r] - mn);    // base-2 domain: native v_exp_f32
      m_i[mi][r] = mn;
      p0[r] = exp2f(s0 - mn);
      p1[r] = exp2f(s1 - mn);
      float rs = p0[r] + p1[r];
      rs += __shfl_xor(rs, 1);
      rs += __shfl_xor(rs, 2);
      rs += __shfl_xor(rs, 4);
      rs += __shfl_xor(rs, 8);
      l_i[mi][r] = l_i[mi][r] * alpha[r] + rs;
    }
#pragma unroll
    for (int j = 0; j < 4; ++j)
#pragma unroll
      for (int r = 0; r < 8; ++r) o[mi][j][r] *= alpha[r];

    // stage P rows (16 x 32 keys bf16) through LDS: C-layout -> A-layout
#pragma unroll
    for (int r = 0; r < 8; ++r) {
      const int row = 16 * mi + r + half * 8;
      Pw[row * 32 + ln]      = f2bf(p0[r]);
      Pw[row * 32 + 16 + ln] = f2bf(p1[r]);
    }
  }
  asm volatile("s_wait_dscnt 0" ::: "memory");

  v16bf pa[2];
#pragma unroll
  for (int mi = 0; mi < 2; ++mi) {
    const unsigned short* pp = Pw + (16 * mi + ln) * 32 + half * 8;
    *((uint4*)&pa[mi])     = *(const uint4*)(pp);
    *((uint4*)&pa[mi] + 1) = *(const uint4*)(pp + 16);
  }

  // PV: V fragments reused across both row tiles
  const int krow = kb + half * 8;
#pragma unroll
  for (int j = 0; j < 4; ++j) {
    const unsigned short* vp = Vh + (size_t)(16 * j + ln) * T_SEQ + krow;
    v16bf vb;
    *((uint4*)&vb)     = *(const uint4*)(vp);
    *((uint4*)&vb + 1) = *(const uint4*)(vp + 16);
    o[0][j] = wmma_bf16(pa[0], vb, o[0][j]);
    o[1][j] = wmma_bf16(pa[1], vb, o[1][j]);
  }
}

// ---------------- flash attention (per-wave 32 query rows) ----------------
__global__ __launch_bounds__(128) void attn_flash(
    const unsigned short* __restrict__ Q, const unsigned short* __restrict__ K,
    const unsigned short* __restrict__ Vt, unsigned short* __restrict__ AO) {
  const int lane = threadIdx.x & 31;
  const int w    = threadIdx.x >> 5;
  const int ln   = lane & 15;
  const int half = lane >> 4;
  const int head  = blockIdx.y;
  const int qtile = blockIdx.x * 128 + w * 32;

  __shared__ unsigned short Pld[4][32 * 32];
  unsigned short* Pw = Pld[w];

  const unsigned short* Qh = Q  + (size_t)head * T_SEQ * HEAD_DIM;
  const unsigned short* Kh = K  + (size_t)head * T_SEQ * HEAD_DIM;
  const unsigned short* Vh = Vt + (size_t)head * HEAD_DIM * T_SEQ;

  v16bf qa[2][2];
#pragma unroll
  for (int mi = 0; mi < 2; ++mi) {
    qa[mi][0] = load_frag_a(Qh, HEAD_DIM, qtile + 16 * mi, 0,  lane);
    qa[mi][1] = load_frag_a(Qh, HEAD_DIM, qtile + 16 * mi, 32, lane);
  }

  float m_i[2][8], l_i[2][8];
  v8f o[2][4];
#pragma unroll
  for (int mi = 0; mi < 2; ++mi) {
#pragma unroll
    for (int r = 0; r < 8; ++r) { m_i[mi][r] = -3.0e38f; l_i[mi][r] = 0.0f; }
#pragma unroll
    for (int j = 0; j < 4; ++j) o[mi][j] = (v8f){};
  }

  // interior tiles: every key visible to every row -> no masking work
  for (int kb = 0; kb < qtile; kb += 32) {
    __builtin_prefetch(Kh + (size_t)(kb + 32 + lane) * HEAD_DIM, 0, 0);
    __builtin_prefetch(Vh + (size_t)lane * T_SEQ + kb + 32, 0, 0);
    attn_tile<false>(Kh, Vh, Pw, qa, o, m_i, l_i, kb, qtile, lane, ln, half);
  }
  // diagonal tile: causal mask applies
  attn_tile<true>(Kh, Vh, Pw, qa, o, m_i, l_i, qtile, qtile, lane, ln, half);

  // finalize: divide by l, merge heads into AO[T][C] bf16
#pragma unroll
  for (int mi = 0; mi < 2; ++mi)
#pragma unroll
    for (int r = 0; r < 8; ++r) {
      const float inv = 1.0f / l_i[mi][r];
      const size_t row =
          (size_t)(qtile + 16 * mi + r + half * 8) * C_DIM + head * HEAD_DIM;
      AO[row + 0  + ln] = f2bf(o[mi][0][r] * inv);
      AO[row + 16 + ln] = f2bf(o[mi][1][r] * inv);
      AO[row + 32 + ln] = f2bf(o[mi][2][r] * inv);
      AO[row + 48 + ln] = f2bf(o[mi][3][r] * inv);
    }
}

// ---------------- output projection GEMM (64x64 per wave) ----------------
__global__ __launch_bounds__(128) void proj_gemm(
    const unsigned short* __restrict__ AO, const unsigned short* __restrict__ WpT,
    const float* __restrict__ b_proj, float* __restrict__ out) {
  const int lane = threadIdx.x & 31;
  const int w    = threadIdx.x >> 5;
  const int ln   = lane & 15;
  const int half = lane >> 4;
  const int ntile = blockIdx.x * 256 + w * 64;
  const int mtile = blockIdx.y * 64;

  v8f acc[4][4];
#pragma unroll
  for (int mi = 0; mi < 4; ++mi)
#pragma unroll
    for (int ni = 0; ni < 4; ++ni) acc[mi][ni] = (v8f){};

  for (int kk = 0; kk < C_DIM; kk += 32) {
    v16bf a[4];
#pragma unroll
    for (int mi = 0; mi < 4; ++mi)
      a[mi] = load_frag_a(AO, C_DIM, mtile + 16 * mi, kk, lane);
#pragma unroll
    for (int ni = 0; ni < 4; ++ni) {
      v16bf b = load_frag_b(WpT, C_DIM, ntile + 16 * ni, kk, lane);
#pragma unroll
      for (int mi = 0; mi < 4; ++mi)
        acc[mi][ni] = wmma_bf16(a[mi], b, acc[mi][ni]);
    }
  }

#pragma unroll
  for (int ni = 0; ni < 4; ++ni) {
    const int c = ntile + 16 * ni + ln;
    const float bias = b_proj[c];
#pragma unroll
    for (int mi = 0; mi < 4; ++mi) {
      const int rowbase = mtile + 16 * mi + half * 8;
#pragma unroll
      for (int r = 0; r < 8; ++r)
        out[(size_t)(rowbase + r) * C_DIM + c] = acc[mi][ni][r] + bias;
    }
  }
}

// ---------------- launch ----------------
extern "C" void kernel_launch(void* const* d_in, const int* in_sizes, int n_in,
                              void* d_out, int out_size, void* d_ws, size_t ws_size,
                              hipStream_t stream) {
  const float* x      = (const float*)d_in[0];
  const float* W_attn = (const float*)d_in[1];
  const float* b_attn = (const float*)d_in[2];
  const float* W_proj = (const float*)d_in[3];
  const float* b_proj = (const float*)d_in[4];
  float* out = (float*)d_out;

  char* ws = (char*)d_ws;
  const size_t MB = (size_t)1 << 20;
  unsigned short* xh  = (unsigned short*)(ws + 0);        //  8 MiB: x bf16 [4096][1024]
  unsigned short* WaT = (unsigned short*)(ws + 8  * MB);  //  6 MiB: W_attn^T bf16 [3072][1024]
  unsigned short* WpT = (unsigned short*)(ws + 14 * MB);  //  2 MiB: W_proj^T bf16 [1024][1024]
  unsigned short* Qs  = (unsigned short*)(ws + 16 * MB);  //  8 MiB: Q bf16 [16][4096][64] (pre-scaled, log2 domain)
  unsigned short* Ks  = (unsigned short*)(ws + 24 * MB);  //  8 MiB: K bf16 [16][4096][64]
  unsigned short* Vt  = (unsigned short*)(ws + 32 * MB);  //  8 MiB: V bf16 [16][64][4096]
  unsigned short* AO  = (unsigned short*)(ws + 40 * MB);  //  8 MiB: attn out bf16 [4096][1024]

  // conversions
  {
    int n = T_SEQ * C_DIM;
    cvt_f32_bf16<<<(n + 255) / 256, 256, 0, stream>>>(x, xh, n);
    int na = C_DIM * QKV_N;
    tpose_f32_bf16<<<(na + 255) / 256, 256, 0, stream>>>(W_attn, WaT, C_DIM, QKV_N);
    int np = C_DIM * C_DIM;
    tpose_f32_bf16<<<(np + 255) / 256, 256, 0, stream>>>(W_proj, WpT, C_DIM, C_DIM);
  }

  // QKV projection: block tile 64M x 256N
  qkv_gemm<<<dim3(QKV_N / 256, T_SEQ / 64), 128, 0, stream>>>(xh, WaT, b_attn, Qs, Ks, Vt);

  // attention: 128 query rows per block (32 per wave), 16 heads
  attn_flash<<<dim3(T_SEQ / 128, N_HEAD), 128, 0, stream>>>(Qs, Ks, Vt, AO);

  // output projection: block tile 64M x 256N
  proj_gemm<<<dim3(C_DIM / 256, T_SEQ / 64), 128, 0, stream>>>(AO, WpT, b_proj, out);
}
